// PD4HG_68143951119112
// MI455X (gfx1250) — matrix operand order repaired and checked by reading, not compiled, not run on hardware
//
#include <hip/hip_runtime.h>
#include <math.h>

typedef __attribute__((ext_vector_type(16))) _Float16 v16h;
typedef __attribute__((ext_vector_type(8)))  _Float16 v8h;
typedef __attribute__((ext_vector_type(8)))  float    v8f;

#define B_N   4096
#define C_N   512
#define F_N   128
#define K_N   256        // fused K: [0..127]=term1, [128..255]=term2
#define TAU_F 32.0f
#define ALPHA_F 0.9f

// ---------------- class prep: Bmat[c][0..127]=exp(-clip(lv)), [128..255]=-2*mu*inv; t3[c]=sum(mu^2*inv)
__global__ void cls_prep(const float* __restrict__ means,
                         const float* __restrict__ log_vars,
                         _Float16* __restrict__ Bmat,
                         float* __restrict__ t3) {
  __shared__ float sbuf[F_N];
  const int c = blockIdx.x, f = threadIdx.x;
  const float m  = means[c * F_N + f];
  float lv = log_vars[c * F_N + f];
  lv = fminf(fmaxf(lv, 0.0f), 6.0f);
  const float inv = __expf(-lv);

  sbuf[f] = m * m;
  __syncthreads();
  for (int s = F_N / 2; s > 0; s >>= 1) {
    if (f < s) sbuf[f] += sbuf[f + s];
    __syncthreads();
  }
  const float nrm = sqrtf(sbuf[0]);
  __syncthreads();
  const float mu = m / fmaxf(nrm, 1e-12f);

  sbuf[f] = mu * mu * inv;
  __syncthreads();
  for (int s = F_N / 2; s > 0; s >>= 1) {
    if (f < s) sbuf[f] += sbuf[f + s];
    __syncthreads();
  }
  if (f == 0) t3[c] = sbuf[0];

  Bmat[(size_t)c * K_N + f]       = (_Float16)inv;
  Bmat[(size_t)c * K_N + F_N + f] = (_Float16)(-2.0f * mu * inv);
}

// ---------------- row prep: Afeat[b][0..127]=Xn^2, [128..255]=Xn
__global__ void row_prep(const float* __restrict__ X,
                         _Float16* __restrict__ Afeat) {
  __shared__ float sbuf[F_N];
  const int b = blockIdx.x, f = threadIdx.x;
  const float x = X[(size_t)b * F_N + f];
  sbuf[f] = x * x;
  __syncthreads();
  for (int s = F_N / 2; s > 0; s >>= 1) {
    if (f < s) sbuf[f] += sbuf[f + s];
    __syncthreads();
  }
  const float xn = x / fmaxf(sqrtf(sbuf[0]), 1e-12f);
  Afeat[(size_t)b * K_N + f]       = (_Float16)(xn * xn);
  Afeat[(size_t)b * K_N + F_N + f] = (_Float16)xn;
}

// ---------------- main: one block = 16 rows x 512 cols; 8 waves, each 16x64 via WMMA f16
__global__ __launch_bounds__(256)
void maha_main(const _Float16* __restrict__ Afeat,
               const _Float16* __restrict__ Bmat,
               const float* __restrict__ t3,
               const int* __restrict__ T,
               float* __restrict__ Sout,      // d_out + 1, row-major 4096x512
               float* __restrict__ partials)  // [gridDim.x][2] = {sum log P, count}
{
  __shared__ float redMin[8][16];
  __shared__ float rowMin[16];
  __shared__ float redNum[8][16];
  __shared__ float redDen[8][16];
  __shared__ float rowLC[16][2];

  const int tid  = threadIdx.x;
  const int wave = tid >> 5;
  const int lane = tid & 31;
  const int hl   = lane & 15;     // row (A) or column (B/D) within 16
  const int hi   = lane >> 4;     // half-wave selector
  const int mBase = blockIdx.x * 16;
  const int nWave = wave * 64;

  v8f acc[4];
  #pragma unroll
  for (int t = 0; t < 4; ++t) acc[t] = (v8f){};

  const _Float16* Arow = Afeat + (size_t)(mBase + hl) * K_N;

  #pragma unroll
  for (int ks = 0; ks < 8; ++ks) {
    const int kA = ks * 32;
    // A fragment: elems e <-> K = kA + hi*16 + e (32B contiguous)
    v16h a;
    {
      a = *(const v16h*)(Arow + kA + hi * 16);
    }
    // B fragments: lane holds column n = nWave + t*16 + hl; elems e <-> K = kA + hi*16 + e (32B contiguous)
    const int kB = kA + hi * 16;
    #pragma unroll
    for (int t = 0; t < 4; ++t) {
      const v16h b = *(const v16h*)(Bmat + (size_t)(nWave + t * 16 + hl) * K_N + kB);
      acc[t] = __builtin_amdgcn_wmma_f32_16x16x32_f16(
          false, a, false, b, (short)0, acc[t], false, false);
    }
  }

  // add t3 bias (per column n)
  #pragma unroll
  for (int t = 0; t < 4; ++t) {
    const float bias = t3[nWave + t * 16 + hl];
    #pragma unroll
    for (int r = 0; r < 8; ++r) acc[t][r] += bias;
  }

  // ---- row-min of D (softmax stabilizer): per-lane min over tiles, butterfly over 16 lanes
  float mn[8];
  #pragma unroll
  for (int r = 0; r < 8; ++r)
    mn[r] = fminf(fminf(acc[0][r], acc[1][r]), fminf(acc[2][r], acc[3][r]));
  #pragma unroll
  for (int off = 1; off < 16; off <<= 1) {
    #pragma unroll
    for (int r = 0; r < 8; ++r)
      mn[r] = fminf(mn[r], __shfl_xor(mn[r], off, 32));
  }
  if (hl == 0) {
    #pragma unroll
    for (int r = 0; r < 8; ++r) redMin[wave][hi * 8 + r] = mn[r];
  }
  __syncthreads();
  if (tid < 16) {
    float v = redMin[0][tid];
    #pragma unroll
    for (int w = 1; w < 8; ++w) v = fminf(v, redMin[w][tid]);
    rowMin[tid] = v;
  }
  __syncthreads();

  // ---- pass 2: softmax sums + S output
  float num[8], den[8];
  #pragma unroll
  for (int r = 0; r < 8; ++r) { num[r] = 0.0f; den[r] = 0.0f; }

  #pragma unroll
  for (int t = 0; t < 4; ++t) {
    const int n = nWave + t * 16 + hl;
    #pragma unroll
    for (int r = 0; r < 8; ++r) {
      const int m   = hi * 8 + r;
      const int row = mBase + m;
      const float d = acc[t][r];
      const int tv  = T[(size_t)row * C_N + n];
      const float w = __expf(TAU_F * (rowMin[m] - d));
      den[r] += w;
      float s;
      if (tv) { num[r] += w; s = 1.0f; }
      else    { s = __expf(-ALPHA_F * d); }
      Sout[(size_t)row * C_N + n] = s;
    }
  }

  #pragma unroll
  for (int off = 1; off < 16; off <<= 1) {
    #pragma unroll
    for (int r = 0; r < 8; ++r) {
      num[r] += __shfl_xor(num[r], off, 32);
      den[r] += __shfl_xor(den[r], off, 32);
    }
  }
  if (hl == 0) {
    #pragma unroll
    for (int r = 0; r < 8; ++r) {
      redNum[wave][hi * 8 + r] = num[r];
      redDen[wave][hi * 8 + r] = den[r];
    }
  }
  __syncthreads();
  if (tid < 16) {
    float N = 0.0f, D = 0.0f;
    #pragma unroll
    for (int w = 0; w < 8; ++w) { N += redNum[w][tid]; D += redDen[w][tid]; }
    float lg = 0.0f, cnt = 0.0f;
    if (N > 0.0f) { lg = __logf(N) - __logf(D); cnt = 1.0f; }
    rowLC[tid][0] = lg; rowLC[tid][1] = cnt;
  }
  __syncthreads();
  if (tid == 0) {
    float sl = 0.0f, sc = 0.0f;
    #pragma unroll
    for (int i = 0; i < 16; ++i) { sl += rowLC[i][0]; sc += rowLC[i][1]; }
    partials[blockIdx.x * 2 + 0] = sl;
    partials[blockIdx.x * 2 + 1] = sc;
  }
}

// ---------------- finalize: fixed-order tree reduce of 256 partials
__global__ void finalize(const float* __restrict__ partials, float* __restrict__ out) {
  __shared__ float sl[256], sc[256];
  const int t = threadIdx.x;
  sl[t] = partials[t * 2 + 0];
  sc[t] = partials[t * 2 + 1];
  __syncthreads();
  for (int s = 128; s > 0; s >>= 1) {
    if (t < s) { sl[t] += sl[t + s]; sc[t] += sc[t + s]; }
    __syncthreads();
  }
  if (t == 0) out[0] = -sl[0] / fmaxf(sc[0], 1.0f);
}

extern "C" void kernel_launch(void* const* d_in, const int* in_sizes, int n_in,
                              void* d_out, int out_size, void* d_ws, size_t ws_size,
                              hipStream_t stream) {
  const float* X        = (const float*)d_in[0];   // 4096x128
  const int*   T        = (const int*)  d_in[1];   // 4096x512
  const float* means    = (const float*)d_in[2];   // 512x128
  const float* log_vars = (const float*)d_in[3];   // 512x128
  float* out = (float*)d_out;                      // [loss(1), S(4096*512)]

  char* ws = (char*)d_ws;
  _Float16* Bmat  = (_Float16*)(ws);                // 512*256*2  = 256 KB
  _Float16* Afeat = (_Float16*)(ws + 262144);       // 4096*256*2 = 2 MB
  float*    t3    = (float*)(ws + 2359296);         // 512*4      = 2 KB
  float*    parts = (float*)(ws + 2361344);         // 256*2*4    = 2 KB

  cls_prep <<<C_N, F_N, 0, stream>>>(means, log_vars, Bmat, t3);
  row_prep <<<B_N, F_N, 0, stream>>>(X, Afeat);
  maha_main<<<B_N / 16, 256, 0, stream>>>(Afeat, Bmat, t3, T, out + 1, parts);
  finalize <<<1, 256, 0, stream>>>(parts, out);
}
